// BallActor_88673894793690
// MI455X (gfx1250) — compile-verified, matrix operand order checked by reading, not run on hardware
//
#include <hip/hip_runtime.h>
#include <math.h>

typedef _Float16 half_t;
typedef __attribute__((ext_vector_type(16))) _Float16 v16h;
typedef __attribute__((ext_vector_type(8)))  float    v8f;

#define BS 1024
#define NOBJ 21
#define NNODE (BS*NOBJ)   // 21504

// Branch-free tanh: tanh(x) = 1 - 2/(exp(2x)+1), via native v_exp_f32 / v_rcp_f32.
// Clamp keeps exp finite; |err| ~ 1e-6, far below the f16 quantization used downstream.
__device__ __forceinline__ float fast_tanh(float x) {
    float cx = fminf(fmaxf(x, -9.0f), 9.0f);
    float e  = __builtin_amdgcn_exp2f(cx * 2.88539008177792681f);   // e^(2x)
    return 1.0f - 2.0f * __builtin_amdgcn_rcpf(e + 1.0f);
}
__device__ __forceinline__ float fast_exp(float x) {
    return __builtin_amdgcn_exp2f(x * 1.44269504088896341f);
}

// K-offset within a 32-wide K tile for the 16-bit A/B WMMA fragment layout
// (CDNA5 ISA §7.12.2): lanes 0-15 cover K={0..7,16..23}, lanes 16-31
// cover K={8..15,24..31}; VGPR v holds K pair (2v or 16+2(v-4)) + lane-half offset.
__device__ __forceinline__ int koff(int lane, int h) {
    int v = h >> 1;
    int kk = (v < 4) ? (2 * v) : (16 + 2 * (v - 4));
    return ((lane & 16) ? 8 : 0) + kk + (h & 1);
}

// ---------------------------------------------------------------------------
// Kernel 1: one-time prep.
//   class_tab[3][64] = tanh(emb_table) @ We + be
//   ws2f / wpqf / wm2f / wa1f: f16 weights pre-swizzled into B-fragment order:
//     frag id = k_tile*NT + n_tile; within frag: [lane][16 halves] contiguous.
//   Wpq cols 0..127 = Wm1[0:192]-Wm1[192:384] (P), cols 128..255 = Wm1[192:384] (Q).
// ---------------------------------------------------------------------------
__global__ void prep_kernel(const float* __restrict__ emb, const float* __restrict__ We,
                            const float* __restrict__ be,  const float* __restrict__ Ws2,
                            const float* __restrict__ Wm1, const float* __restrict__ Wm2,
                            const float* __restrict__ Wa1,
                            float* __restrict__ class_tab, half_t* __restrict__ ws2f,
                            half_t* __restrict__ wpqf, half_t* __restrict__ wm2f,
                            half_t* __restrict__ wa1f)
{
    int idx = blockIdx.x * blockDim.x + threadIdx.x;
    if (idx < 192) {
        int c = idx >> 6, n = idx & 63;
        float acc = be[n];
        for (int e = 0; e < 64; ++e) acc += fast_tanh(emb[c * 64 + e]) * We[e * 64 + n];
        class_tab[idx] = acc;
        return;
    }
    idx -= 192;
    if (idx < 16384) {                       // Ws2 (128x128), NT=8, KT=4
        int frag = idx >> 9, rem = idx & 511, lane = rem >> 4, h = rem & 15;
        int kt = frag >> 3, nt = frag & 7;
        int n = nt * 16 + (lane & 15);
        int k = kt * 32 + koff(lane, h);
        ws2f[idx] = (half_t)Ws2[k * 128 + n];
        return;
    }
    idx -= 16384;
    if (idx < 49152) {                       // Wpq (192x256), NT=16, KT=6
        int frag = idx >> 9, rem = idx & 511, lane = rem >> 4, h = rem & 15;
        int kt = frag >> 4, nt = frag & 15;
        int n = nt * 16 + (lane & 15);
        int k = kt * 32 + koff(lane, h);
        float v;
        if (n < 128) v = Wm1[k * 128 + n] - Wm1[(192 + k) * 128 + n];
        else         v = Wm1[(192 + k) * 128 + (n - 128)];
        wpqf[idx] = (half_t)v;
        return;
    }
    idx -= 49152;
    if (idx < 16384) {                       // Wm2 (128x128), NT=8, KT=4
        int frag = idx >> 9, rem = idx & 511, lane = rem >> 4, h = rem & 15;
        int kt = frag >> 3, nt = frag & 7;
        int n = nt * 16 + (lane & 15);
        int k = kt * 32 + koff(lane, h);
        wm2f[idx] = (half_t)Wm2[k * 128 + n];
        return;
    }
    idx -= 16384;
    if (idx < 16384) {                       // Wa1 (128x128), NT=8, KT=4
        int frag = idx >> 9, rem = idx & 511, lane = rem >> 4, h = rem & 15;
        int kt = frag >> 3, nt = frag & 7;
        int n = nt * 16 + (lane & 15);
        int k = kt * 32 + koff(lane, h);
        wa1f[idx] = (half_t)Wa1[k * 128 + n];
    }
}

// ---------------------------------------------------------------------------
// Kernel 2: node features + fused P/Q GEMM.  One wave32 per 16 nodes.
//   h1 = tanh(spatial_inp @ Ws1 + bs1)                  (scalar, K=4)
//   spatial_embed = h1 @ Ws2 + bs2                      (WMMA 16x128x128)
//   feat = tanh([spatial_embed, class_tab[cat]])        (f16, LDS)
//   [P|Q] = feat @ Wpq (+bm1 on P cols)                 (WMMA 16x256x192)
// ---------------------------------------------------------------------------
__global__ void __launch_bounds__(32) feat_pq_kernel(
    const float* __restrict__ state_inp, const float* __restrict__ tar_scores,
    const float* __restrict__ class_tab, const float* __restrict__ Ws1,
    const float* __restrict__ bs1, const float* __restrict__ bs2,
    const half_t* __restrict__ ws2f, const half_t* __restrict__ wpqf,
    const float* __restrict__ bm1, float* __restrict__ pqbuf)
{
    __shared__ half_t h1f[16 * 128];
    __shared__ half_t featf[16 * 192];
    int l = threadIdx.x;
    int nb0 = blockIdx.x * 16;

    {   // each lane: one node (m = l/2), half of the 128 hidden channels
        int m = l >> 1;
        int node = nb0 + m;
        int b = node / NOBJ, o = node % NOBJ;
        float in0 = state_inp[b * 63 + o * 3 + 0];
        float in1 = state_inp[b * 63 + o * 3 + 1];
        float in2 = fast_tanh(tar_scores[node * 2 + 0]);
        float in3 = fast_tanh(tar_scores[node * 2 + 1]);
        int hbase = (l & 1) * 64;
        for (int hh = 0; hh < 64; ++hh) {
            int h = hbase + hh;
            float acc = in0 * Ws1[h] + in1 * Ws1[128 + h] + in2 * Ws1[256 + h]
                      + in3 * Ws1[384 + h] + bs1[h];
            h1f[m * 128 + h] = (half_t)fast_tanh(acc);
        }
        int cat = (int)state_inp[b * 63 + o * 3 + 2];
        int ebase = (l & 1) * 32;
        for (int e = 0; e < 32; ++e)
            featf[m * 192 + 128 + ebase + e] = (half_t)fast_tanh(class_tab[cat * 64 + ebase + e]);
    }
    __syncthreads();

    // spatial_embed via WMMA
    for (int nt = 0; nt < 8; ++nt) {
        int n = nt * 16 + (l & 15);
        float bias = bs2[n];
        v8f c; for (int r = 0; r < 8; ++r) c[r] = bias;
#pragma unroll
        for (int kt = 0; kt < 4; ++kt) {
            v16h a;
            int m = l & 15;
            for (int h = 0; h < 16; ++h)
                a[h] = h1f[m * 128 + kt * 32 + koff(l, h)];
            v16h bf = *(const v16h*)(ws2f + ((kt * 8 + nt) * 32 + l) * 16);
            c = __builtin_amdgcn_wmma_f32_16x16x32_f16(false, a, false, bf, (short)0, c, false, false);
        }
        for (int r = 0; r < 8; ++r) {
            int m = r + ((l & 16) ? 8 : 0);
            featf[m * 192 + n] = (half_t)fast_tanh(c[r]);
        }
    }
    __syncthreads();

    // [P|Q] via WMMA (K=192, N=256)
    for (int nt = 0; nt < 16; ++nt) {
        int n = nt * 16 + (l & 15);
        float bias = (nt < 8) ? bm1[n] : 0.0f;
        v8f c; for (int r = 0; r < 8; ++r) c[r] = bias;
#pragma unroll
        for (int kt = 0; kt < 6; ++kt) {
            v16h a;
            int m = l & 15;
            for (int h = 0; h < 16; ++h)
                a[h] = featf[m * 192 + kt * 32 + koff(l, h)];
            v16h bf = *(const v16h*)(wpqf + ((kt * 16 + nt) * 32 + l) * 16);
            c = __builtin_amdgcn_wmma_f32_16x16x32_f16(false, a, false, bf, (short)0, c, false, false);
        }
        for (int r = 0; r < 8; ++r) {
            int m = r + ((l & 16) ? 8 : 0);
            pqbuf[(nb0 + m) * 256 + n] = c[r];
        }
    }
}

// ---------------------------------------------------------------------------
// Kernel 3: edge MLP (max-pooled) + actor head. One block (4 waves) per batch.
//   msg = tanh(P_i + Q_j) @ Wm2 + bm2 ; x_i = tanh(max_j msg)   (WMMA)
//   a1  = tanh(x @ Wa1 + ba1)                                    (WMMA)
//   act = a1 @ Wa2 + ba2 -> mu, std                              (scalar)
// ---------------------------------------------------------------------------
__global__ void __launch_bounds__(128) edge_actor_kernel(
    const float* __restrict__ pqbuf, const half_t* __restrict__ wm2f,
    const float* __restrict__ bm2, const half_t* __restrict__ wa1f,
    const float* __restrict__ ba1, const float* __restrict__ Wa2,
    const float* __restrict__ ba2, float* __restrict__ out)
{
    __shared__ float  PQ[NOBJ * 256];   // per node: P[0..127], Q[128..255]
    __shared__ half_t XBh[32 * 128];    // pooled node features, zero-padded to 32 rows
    __shared__ float  A1[NOBJ * 128];
    int t = threadIdx.x;
    int b = blockIdx.x;
    int w = t >> 5, l = t & 31;

    for (int idx = t; idx < NOBJ * 256; idx += 128)
        PQ[idx] = pqbuf[b * NOBJ * 256 + idx];
    for (int idx = t; idx < 32 * 128; idx += 128)
        XBh[idx] = (half_t)0.0f;        // zero pad rows 21..31 (and init the rest)
    __syncthreads();

    for (int i = w; i < NOBJ; i += 4) {
        const float* Pi = &PQ[i * 256];
        float cm[8];
        for (int nt = 0; nt < 8; ++nt) cm[nt] = -INFINITY;

        for (int mt = 0; mt < 2; ++mt) {
            // Build A fragments: rows = edges j' (20 valid, pad to 32 with zeros)
            int mrow = mt * 16 + (l & 15);
            bool valid = (mrow < 20);
            int j = valid ? (mrow + (mrow >= i ? 1 : 0)) : 0;
            const float* Qj = &PQ[j * 256 + 128];
            v16h af[4];
#pragma unroll
            for (int kt = 0; kt < 4; ++kt)
                for (int h = 0; h < 16; ++h) {
                    int k = kt * 32 + koff(l, h);
                    af[kt][h] = valid ? (half_t)fast_tanh(Pi[k] + Qj[k]) : (half_t)0.0f;
                }

            for (int nt = 0; nt < 8; ++nt) {
                int n = nt * 16 + (l & 15);
                float bias = bm2[n];
                v8f c; for (int r = 0; r < 8; ++r) c[r] = bias;
#pragma unroll
                for (int kt = 0; kt < 4; ++kt) {
                    v16h bf = *(const v16h*)(wm2f + ((kt * 8 + nt) * 32 + l) * 16);
                    c = __builtin_amdgcn_wmma_f32_16x16x32_f16(false, af[kt], false, bf, (short)0, c, false, false);
                }
                // row-max over valid edge rows of this tile
                float mx;
                if (mt == 0) {                      // rows 0..15 all valid
                    mx = c[0];
                    for (int r = 1; r < 8; ++r) mx = fmaxf(mx, c[r]);
                } else {                            // rows 16..19 valid (lanes<16, r<4)
                    if (l < 16) { mx = c[0]; for (int r = 1; r < 4; ++r) mx = fmaxf(mx, c[r]); }
                    else mx = -INFINITY;
                }
                mx = fmaxf(mx, __shfl_xor(mx, 16, 32));  // combine lane halves (same n)
                cm[nt] = fmaxf(cm[nt], mx);
            }
        }
        if (l < 16)
            for (int nt = 0; nt < 8; ++nt)
                XBh[i * 128 + nt * 16 + l] = (half_t)fast_tanh(cm[nt]);
    }
    __syncthreads();

    // a1 = tanh(XB @ Wa1 + ba1) via WMMA: 2 m-tiles x 8 n-tiles, 16 tile-jobs / 4 waves
    for (int job = w; job < 16; job += 4) {
        int mt = job >> 3, nt = job & 7;
        int n = nt * 16 + (l & 15);
        float bias = ba1[n];
        v8f c; for (int r = 0; r < 8; ++r) c[r] = bias;
#pragma unroll
        for (int kt = 0; kt < 4; ++kt) {
            v16h a;
            int m = mt * 16 + (l & 15);
            for (int h = 0; h < 16; ++h)
                a[h] = XBh[m * 128 + kt * 32 + koff(l, h)];
            v16h bf = *(const v16h*)(wa1f + ((kt * 8 + nt) * 32 + l) * 16);
            c = __builtin_amdgcn_wmma_f32_16x16x32_f16(false, a, false, bf, (short)0, c, false, false);
        }
        for (int r = 0; r < 8; ++r) {
            int m = mt * 16 + r + ((l & 16) ? 8 : 0);
            if (m < NOBJ) A1[m * 128 + n] = fast_tanh(c[r]);
        }
    }
    __syncthreads();

    // act = a1 @ Wa2 + ba2 -> mu (cols 0,1), std (cols 2,3)
    for (int idx = t; idx < NOBJ * 4; idx += 128) {
        int i = idx >> 2, c = idx & 3;
        float acc = ba2[c];
        for (int k = 0; k < 128; ++k) acc += A1[i * 128 + k] * Wa2[k * 4 + c];
        if (c < 2) {
            out[b * 42 + i * 2 + c] = 0.3f * fast_tanh(acc);
        } else {
            float ls = fast_tanh(acc);
            ls = -5.0f + 3.5f * (ls + 1.0f);
            out[43008 + b * 42 + i * 2 + (c - 2)] = fast_exp(ls);
        }
    }
}

extern "C" void kernel_launch(void* const* d_in, const int* in_sizes, int n_in,
                              void* d_out, int out_size, void* d_ws, size_t ws_size,
                              hipStream_t stream)
{
    const float* state_inp  = (const float*)d_in[0];
    const float* tar_scores = (const float*)d_in[1];
    const float* emb_table  = (const float*)d_in[2];
    const float* We  = (const float*)d_in[3];
    const float* be  = (const float*)d_in[4];
    const float* Ws1 = (const float*)d_in[5];
    const float* bs1 = (const float*)d_in[6];
    const float* Ws2 = (const float*)d_in[7];
    const float* bs2 = (const float*)d_in[8];
    const float* Wm1 = (const float*)d_in[9];
    const float* bm1 = (const float*)d_in[10];
    const float* Wm2 = (const float*)d_in[11];
    const float* bm2 = (const float*)d_in[12];
    const float* Wa1 = (const float*)d_in[13];
    const float* ba1 = (const float*)d_in[14];
    const float* Wa2 = (const float*)d_in[15];
    const float* ba2 = (const float*)d_in[16];
    float* out = (float*)d_out;

    char* ws = (char*)d_ws;
    float*  class_tab = (float*) (ws + 0);        //    768 B (pad 1024)
    half_t* ws2f      = (half_t*)(ws + 1024);     //  32768 B
    half_t* wpqf      = (half_t*)(ws + 33792);    //  98304 B
    half_t* wm2f      = (half_t*)(ws + 132096);   //  32768 B
    half_t* wa1f      = (half_t*)(ws + 164864);   //  32768 B
    float*  pqbuf     = (float*) (ws + 197632);   // 21504*256*4 = 22 MB

    prep_kernel<<<385, 256, 0, stream>>>(emb_table, We, be, Ws2, Wm1, Wm2, Wa1,
                                         class_tab, ws2f, wpqf, wm2f, wa1f);
    feat_pq_kernel<<<NNODE / 16, 32, 0, stream>>>(state_inp, tar_scores, class_tab,
                                                  Ws1, bs1, bs2, ws2f, wpqf, bm1, pqbuf);
    edge_actor_kernel<<<BS, 128, 0, stream>>>(pqbuf, wm2f, bm2, wa1f, ba1, Wa2, ba2, out);
}
